// FP8WeightOnlyLinear_18004502905013
// MI455X (gfx1250) — matrix operand order repaired and checked by reading, not compiled, not run on hardware
//
#include <hip/hip_runtime.h>
#include <hip/hip_bf16.h>

typedef __attribute__((ext_vector_type(16))) __bf16 v16bf;
typedef __attribute__((ext_vector_type(8)))  __bf16 v8bf;
typedef __attribute__((ext_vector_type(8)))  float  v8f;
typedef __attribute__((ext_vector_type(4)))  float  v4f;

// ---------------------------------------------------------------------------
// Prologue: dequantize fp32 weight (fp8-valued) -> bf16, once, into d_ws.
// w_bf16[i] = bf16(w[i] * scale).  16.7M elements, 8 per thread.
// ---------------------------------------------------------------------------
__global__ __launch_bounds__(256)
void fp8wol_dequant_bf16(const float* __restrict__ w,
                         const float* __restrict__ scale,
                         __bf16* __restrict__ out,
                         int n)
{
    const float s = scale[0];
    const int i = (blockIdx.x * 256 + threadIdx.x) * 8;
    if (i + 8 <= n) {
        v4f a = *(const v4f*)(w + i);
        v4f b = *(const v4f*)(w + i + 4);
        v8bf o;
        o[0] = (__bf16)(a[0] * s);
        o[1] = (__bf16)(a[1] * s);
        o[2] = (__bf16)(a[2] * s);
        o[3] = (__bf16)(a[3] * s);
        o[4] = (__bf16)(b[0] * s);
        o[5] = (__bf16)(b[1] * s);
        o[6] = (__bf16)(b[2] * s);
        o[7] = (__bf16)(b[3] * s);
        *(v8bf*)(out + i) = o;
    }
}

// ---------------------------------------------------------------------------
// Fragment loaders (wave32 WMMA 16x16x32 bf16 layouts).
// A (16x32, M x K): lane l holds row m=l%16; K=0..15 for
// lanes 0-15 and K=16..31 for lanes 16-31  -> two contiguous 16B loads.
// ---------------------------------------------------------------------------
__device__ __forceinline__ v16bf load_a_frag(const __bf16* __restrict__ row,
                                             int k, int koff8)
{
    v8bf lo = *(const v8bf*)(row + k + koff8);        // K = koff8 .. +7
    v8bf hi = *(const v8bf*)(row + k + koff8 + 8);    // K = koff8+8 .. +15
    return __builtin_shufflevector(lo, hi,
        0, 1, 2, 3, 4, 5, 6, 7, 8, 9, 10, 11, 12, 13, 14, 15);
}

// B (32x16, K x N): lane l holds column n=l%16; lanes 0-15 carry K=0..15,
// lanes 16-31 carry K=16..31 -> 16 contiguous bf16 from W row (two 16B loads).
__device__ __forceinline__ v16bf load_b_frag(const __bf16* __restrict__ row,
                                             int k, int koff16)
{
    v8bf lo = *(const v8bf*)(row + k + koff16);       // K = koff16 .. +7
    v8bf hi = *(const v8bf*)(row + k + koff16 + 8);   // K = koff16+8 .. +15
    return __builtin_shufflevector(lo, hi,
        0, 1, 2, 3, 4, 5, 6, 7, 8, 9, 10, 11, 12, 13, 14, 15);
}

// ---------------------------------------------------------------------------
// GEMM: C[M][N](bf16) = A[M][K](bf16) x Wq[N][K](bf16)^T + bias
// Workgroup: 256 threads = 8 wave32, 256x128 C tile (wave grid 4M x 2N).
// Each wave: 64x64 sub-tile = 4x4 WMMA accumulators (16 WMMA vs 16 b128
// loads per k-step -> 16 MACs/byte of cache traffic).
// ---------------------------------------------------------------------------
__global__ __launch_bounds__(256)
void fp8wol_gemm_wmma_bf16(const __bf16* __restrict__ A,    // [M][K]
                           const __bf16* __restrict__ Wq,   // [N][K]
                           const float*  __restrict__ bias, // [N]
                           __bf16* __restrict__ C,          // [M][N]
                           int M, int N, int K)
{
    const int lane  = threadIdx.x & 31;
    const int wid   = threadIdx.x >> 5;     // 0..7
    const int wm    = wid & 3;              // wave row   (M): 0..3
    const int wn    = wid >> 2;             // wave col   (N): 0..1
    const int row16 = lane & 15;
    const int half  = lane >> 4;            // 0 or 1

    const int mBase = blockIdx.y * 256 + wm * 64;   // wave's M origin
    const int nBase = blockIdx.x * 128 + wn * 64;   // wave's N origin

    // Per-lane row pointers (K-major rows).
    const __bf16* aRow[4];
    const __bf16* bRow[4];
#pragma unroll
    for (int i = 0; i < 4; ++i)
        aRow[i] = A + (long long)(mBase + i * 16 + row16) * K;
#pragma unroll
    for (int j = 0; j < 4; ++j)
        bRow[j] = Wq + (long long)(nBase + j * 16 + row16) * K;

    const int aKoff = half * 16;   // A lane K-chunk start
    const int bKoff = half * 16;   // B lane K-chunk start

    v8f acc[4][4];
#pragma unroll
    for (int i = 0; i < 4; ++i)
#pragma unroll
        for (int j = 0; j < 4; ++j)
            acc[i][j] = (v8f)(0.0f);

#pragma unroll 2
    for (int k = 0; k < K; k += 32) {
        v16bf af[4], bf[4];
#pragma unroll
        for (int i = 0; i < 4; ++i)
            af[i] = load_a_frag(aRow[i], k, aKoff);
#pragma unroll
        for (int j = 0; j < 4; ++j)
            bf[j] = load_b_frag(bRow[j], k, bKoff);

#pragma unroll
        for (int i = 0; i < 4; ++i) {
#pragma unroll
            for (int j = 0; j < 4; ++j) {
                acc[i][j] = __builtin_amdgcn_wmma_f32_16x16x32_bf16(
                    /*neg_a=*/false, af[i],
                    /*neg_b=*/false, bf[j],
                    /*c_mod=*/(short)0, acc[i][j],
                    /*reuse_a=*/false, /*reuse_b=*/false);
            }
        }
    }

    // Epilogue: C/D layout -> VGPR r: M = r + 8*half, N = lane%16.
#pragma unroll
    for (int j = 0; j < 4; ++j) {
        const int ncol = nBase + j * 16 + row16;
        const float bv = bias[ncol];
#pragma unroll
        for (int i = 0; i < 4; ++i) {
            const long long mrow0 = mBase + i * 16 + half * 8;
#pragma unroll
            for (int r = 0; r < 8; ++r) {
                C[(mrow0 + r) * (long long)N + ncol] =
                    (__bf16)(acc[i][j][r] + bv);
            }
        }
    }
}

// ---------------------------------------------------------------------------
// Host-side launch.
// Inputs (setup_inputs order): x[bf16, M*K], weight[f32, N*K],
//                              weight_scale[f32, 1], bias[f32, N]
// Output: bf16, M*N.  d_ws holds the dequantized bf16 weight (N*K*2 bytes).
// ---------------------------------------------------------------------------
extern "C" void kernel_launch(void* const* d_in, const int* in_sizes, int n_in,
                              void* d_out, int out_size, void* d_ws, size_t ws_size,
                              hipStream_t stream)
{
    const __bf16* x     = (const __bf16*)d_in[0];
    const float*  w     = (const float*)d_in[1];
    const float*  scale = (const float*)d_in[2];
    const float*  bias  = (const float*)d_in[3];
    __bf16*       out   = (__bf16*)d_out;

    const int wcount = in_sizes[1];          // N*K
    const int N      = in_sizes[3];          // D_out
    const int K      = wcount / N;           // D_in
    const int M      = in_sizes[0] / K;      // B*S

    __bf16* wq = (__bf16*)d_ws;

    // 1) Dequantize weight -> bf16 in workspace.
    {
        const int elems_per_blk = 256 * 8;
        const int grid = (wcount + elems_per_blk - 1) / elems_per_blk;
        fp8wol_dequant_bf16<<<grid, 256, 0, stream>>>(w, scale, wq, wcount);
    }

    // 2) WMMA GEMM + bias.
    {
        dim3 grid(N / 128, M / 256);
        fp8wol_gemm_wmma_bf16<<<grid, dim3(256), 0, stream>>>(
            x, wq, bias, out, M, N, K);
    }
}
